// NNAttentionHead_24249385353814
// MI455X (gfx1250) — matrix operand order repaired and checked by reading, not compiled, not run on hardware
//
#include <hip/hip_runtime.h>
#include <hip/hip_bf16.h>

// ---------------------------------------------------------------------------
// NN attention head (B=4,T=512,C=128,H=128) for MI455X / gfx1250 (wave32).
//
// Decomposition:  gelu( concat(xk,xq)+pos_emb ) @ W1  ==
//                 gelu( P + XK + XQ ) with P = pos_emb@W1 (batch-independent),
//                 XK = x@W1[:C,:], XQ = x@W1[C:,:].
// P GEMM (17.2 GFLOP) runs on v_wmma_f32_16x16x32_bf16 fused per tile so the
// 268MB pos_emb tensor streams from HBM exactly once (~12us at 23.3 TB/s).
// Final wei@V uses full-precision v_wmma_f32_16x16x4_f32.
//
// Workspace requirement: ~8 MB (asserted by layout below).
// ---------------------------------------------------------------------------

#define B 4
#define T 512
#define C 128
#define H 128

typedef __bf16 bf16_t;
typedef __attribute__((ext_vector_type(16))) __bf16 v16bf;
typedef __attribute__((ext_vector_type(8)))  __bf16 v8bf;
typedef __attribute__((ext_vector_type(8)))  float  v8f;
typedef __attribute__((ext_vector_type(2)))  float  v2f;

#define CAT16(lo, hi) __builtin_shufflevector((lo), (hi), 0,1,2,3,4,5,6,7,8,9,10,11,12,13,14,15)

__device__ __forceinline__ float gelu_exact(float x) {
  return 0.5f * x * (1.0f + erff(x * 0.70710678118654752440f));
}

// ---------------------------------------------------------------------------
// Kernel 1: dtype conversions + weight transposes (bf16 staging buffers).
//   xb  [B*T][C]   = bf16(x)
//   W1T [C][2C]    = bf16(W1^T)   (row n holds K-contiguous data for WMMA B)
//   WvT [H][C]     = bf16(Wv^T)
// ---------------------------------------------------------------------------
__global__ __launch_bounds__(256) void prep_convert(
    const float* __restrict__ x, const float* __restrict__ W1,
    const float* __restrict__ Wv,
    bf16_t* __restrict__ xb, bf16_t* __restrict__ W1T, bf16_t* __restrict__ WvT) {
  const int tid = blockIdx.x * blockDim.x + threadIdx.x;
  const int stride = gridDim.x * blockDim.x;
  for (int i = tid; i < B * T * C; i += stride) xb[i] = (bf16_t)x[i];
  for (int i = tid; i < 2 * C * C; i += stride) {
    int n = i / (2 * C), k = i % (2 * C);
    W1T[i] = (bf16_t)W1[k * C + n];
  }
  for (int i = tid; i < C * H; i += stride) {
    int h = i / C, c = i % C;
    WvT[i] = (bf16_t)Wv[c * H + h];
  }
}

// ---------------------------------------------------------------------------
// Kernel 2: small projections via bf16 WMMA.
//   which=0: XK = x@W1[:C,:]   which=1: XQ = x@W1[C:,:]   which=2: V = x@Wv
// M = B*T = 2048 (16-row tiles), N = 128 (one 16-col tile per wave), K = 128.
// ---------------------------------------------------------------------------
__global__ __launch_bounds__(256) void proj_gemm(
    const bf16_t* __restrict__ xb, const bf16_t* __restrict__ W1T,
    const bf16_t* __restrict__ WvT,
    float* __restrict__ XK, float* __restrict__ XQ, float* __restrict__ Vv) {
  const int w = threadIdx.x >> 5;      // wave id = N-subtile
  const int L = threadIdx.x & 31;
  const int m16 = L & 15, hf = L >> 4;
  const int Mtile = blockIdx.x;        // 0..127
  const int which = blockIdx.z;        // 0:XK 1:XQ 2:V

  const bf16_t* Bmat; int bstride, koff; float* outp;
  if (which == 0)      { Bmat = W1T; bstride = 2 * C; koff = 0; outp = XK; }
  else if (which == 1) { Bmat = W1T; bstride = 2 * C; koff = C; outp = XQ; }
  else                 { Bmat = WvT; bstride = C;     koff = 0; outp = Vv; }

  const int row = Mtile * 16 + m16;    // global M row (b*T + t)
  const int n   = w * 16 + m16;        // output column for this lane

  v8f acc = {};
  #pragma unroll
  for (int k = 0; k < C; k += 32) {
    // A fragment (16x32 bf16): lane=M row; K = k + hf*8 + {0..7}, k+16+hf*8 + {0..7}
    const bf16_t* ap = xb + (size_t)row * C + k + hf * 8;
    v16bf a = CAT16(*(const v8bf*)ap, *(const v8bf*)(ap + 16));
    // B fragment (32x16 bf16): lane=N col; K = k + hf*8 + {0..7}, k+16+hf*8 + {0..7}
    const bf16_t* bp = Bmat + (size_t)n * bstride + koff + k + hf * 8;
    v16bf bb = CAT16(*(const v8bf*)bp, *(const v8bf*)(bp + 16));
    acc = __builtin_amdgcn_wmma_f32_16x16x32_bf16(false, a, false, bb,
                                                  (short)0, acc, false, false);
  }
  #pragma unroll
  for (int r = 0; r < 8; ++r) {
    const int orow = Mtile * 16 + r + hf * 8;   // D layout: vgpr r -> M = r + 8*half
    outp[(size_t)orow * C + n] = acc[r];
  }
}

// ---------------------------------------------------------------------------
// Kernel 3: fused  P = pos_emb_tile @ W1  (bf16 WMMA, tile in LDS)
//           then  logit[b,tq,tk] = (sum_c gelu(P+XK+XQ)*W2[c] + b2)*C^-0.5
// Block tile: 4 tq x 16 tk = 64 GEMM rows, N = 128, K = 256.
// LDS buffer is reused: bf16 A-tile (padded pitch 264) -> fp32 P-tile (pitch 129).
// ---------------------------------------------------------------------------
#define TQT 4
#define TKT 16
#define ROWS (TQT * TKT)          // 64
#define APITCH 264                // ushorts; 528B rows -> conflict-reduced
#define PPITCH 129                // floats;  odd pitch -> conflict-free column reads

__global__ __launch_bounds__(256) void fused_logits(
    const float* __restrict__ pos_emb, const bf16_t* __restrict__ W1T,
    const float* __restrict__ XK, const float* __restrict__ XQ,
    const float* __restrict__ W2, const float* __restrict__ b2p,
    float* __restrict__ WEI) {
  __shared__ __align__(16) unsigned char smem[ROWS * APITCH * 2];  // 33792 B
  __shared__ float red[256];
  bf16_t* A_lds = (bf16_t*)smem;
  float*  P_lds = (float*)smem;   // 64*129*4 = 33024 B <= 33792 B

  const int tid = threadIdx.x;
  const int tk0 = blockIdx.x * TKT;
  const int tq0 = blockIdx.y * TQT;

  // ---- Stage pos_emb tile -> LDS as bf16 (HBM read: exactly once overall) ----
  for (int i = tid; i < ROWS * 64; i += 256) {       // 64 rows x 64 float4
    const int r = i >> 6;
    const int c4 = (i & 63) << 2;
    const int tqi = r >> 4, tki = r & 15;
    const float4 f = *(const float4*)(pos_emb +
        (((size_t)(tq0 + tqi)) * T + (tk0 + tki)) * (2 * C) + c4);
    bf16_t* dst = A_lds + r * APITCH + c4;
    dst[0] = (bf16_t)f.x; dst[1] = (bf16_t)f.y;
    dst[2] = (bf16_t)f.z; dst[3] = (bf16_t)f.w;
  }
  __syncthreads();

  // ---- GEMM: 64x128 tile, K=256; wave w owns N-subtile w, 4 M-subtiles ----
  const int w = tid >> 5, L = tid & 31;
  const int m16 = L & 15, hf = L >> 4;
  const int n = w * 16 + m16;
  v8f acc[TQT] = {};
  #pragma unroll
  for (int k = 0; k < 2 * C; k += 32) {
    const bf16_t* bp = W1T + (size_t)n * (2 * C) + k + hf * 8;
    const v16bf bfrag = CAT16(*(const v8bf*)bp, *(const v8bf*)(bp + 16));
    #pragma unroll
    for (int ms = 0; ms < TQT; ++ms) {
      const bf16_t* ap = A_lds + (ms * 16 + m16) * APITCH + k + hf * 8;
      const v16bf afrag = CAT16(*(const v8bf*)ap, *(const v8bf*)(ap + 16));
      acc[ms] = __builtin_amdgcn_wmma_f32_16x16x32_bf16(false, afrag, false, bfrag,
                                                        (short)0, acc[ms], false, false);
    }
  }
  __syncthreads();                       // everyone done reading A_lds
  #pragma unroll
  for (int ms = 0; ms < TQT; ++ms)
    #pragma unroll
    for (int r = 0; r < 8; ++r)
      P_lds[(ms * 16 + r + hf * 8) * PPITCH + n] = acc[ms][r];
  __syncthreads();

  // ---- GELU + W2 reduction, per batch (4 threads per row, 32 c each) ----
  const int r   = tid & 63;
  const int ch  = tid >> 6;              // 0..3 -> c chunk of 32
  const int tqi = r >> 4, tki = r & 15;
  const int gtq = tq0 + tqi, gtk = tk0 + tki;
  const bool live = (gtk <= gtq);        // causal mask
  const float b2v = b2p[0];
  const float* prow = &P_lds[r * PPITCH + ch * 32];
  const float* w2p  = W2 + ch * 32;

  for (int b = 0; b < B; ++b) {
    float s = 0.0f;
    if (live) {
      const float* xkrow = XK + ((size_t)(b * T + gtk)) * C + ch * 32;
      const float* xqrow = XQ + ((size_t)(b * T + gtq)) * C + ch * 32;
      #pragma unroll 8
      for (int c = 0; c < 32; ++c) {
        const float xv = prow[c] + xkrow[c] + xqrow[c];
        s += gelu_exact(xv) * w2p[c];
      }
    }
    red[tid] = s;
    __syncthreads();
    if (ch == 0) {
      const float tot = red[tid] + red[tid + 64] + red[tid + 128] + red[tid + 192];
      const float logit = (tot + b2v) * 0.08838834764831845f;   // C^-0.5
      WEI[((size_t)b * T + gtq) * T + gtk] = live ? logit : -__builtin_inff();
    }
    __syncthreads();
  }
}

// ---------------------------------------------------------------------------
// Kernel 4: causal softmax over tk for each of the B*T rows.
// ---------------------------------------------------------------------------
__global__ __launch_bounds__(256) void softmax_rows(float* __restrict__ WEI) {
  float* p = WEI + (size_t)blockIdx.x * T;
  const int tid = threadIdx.x;
  __shared__ float sm[256];

  const float a0 = p[tid], a1 = p[tid + 256];
  sm[tid] = fmaxf(a0, a1);
  __syncthreads();
  for (int s = 128; s > 0; s >>= 1) {
    if (tid < s) sm[tid] = fmaxf(sm[tid], sm[tid + s]);
    __syncthreads();
  }
  const float m = sm[0];
  __syncthreads();

  const float e0 = expf(a0 - m), e1 = expf(a1 - m);   // exp(-inf)=0 handles mask
  sm[tid] = e0 + e1;
  __syncthreads();
  for (int s = 128; s > 0; s >>= 1) {
    if (tid < s) sm[tid] += sm[tid + s];
    __syncthreads();
  }
  const float inv = 1.0f / sm[0];
  p[tid] = e0 * inv;
  p[tid + 256] = e1 * inv;
}

// ---------------------------------------------------------------------------
// Kernel 5: out = wei @ V, full fp32 via v_wmma_f32_16x16x4_f32 (K = T = 512).
// One 16x16 output tile per wave; 1024 tiles total.
// ---------------------------------------------------------------------------
__global__ __launch_bounds__(256) void attn_out(
    const float* __restrict__ WEI, const float* __restrict__ Vv,
    float* __restrict__ out) {
  const int w = threadIdx.x >> 5, L = threadIdx.x & 31;
  const int m16 = L & 15, hf = L >> 4;
  const int gw = blockIdx.x * 8 + w;    // 0..1023
  const int b = gw >> 8;
  const int Mtile = (gw >> 3) & 31;
  const int Hsub = gw & 7;

  const float* Arow  = WEI + ((size_t)b * T + Mtile * 16 + m16) * T;
  const float* Bbase = Vv + (size_t)b * T * H + Hsub * 16 + m16;

  v8f acc = {};
  #pragma unroll 4
  for (int k = 0; k < T; k += 4) {
    // A (16x4 f32): lane=M; VGPR j -> K = k + 2*half + j  (contiguous pair)
    const float* ap = Arow + k + hf * 2;
    v2f a; a[0] = ap[0]; a[1] = ap[1];
    // B (4x16 f32): lane=N; VGPR j -> K = k + 2*half + j  (row-strided in V)
    const float* bp = Bbase + (size_t)(k + hf * 2) * H;
    v2f bb; bb[0] = bp[0]; bb[1] = bp[H];
    acc = __builtin_amdgcn_wmma_f32_16x16x4_f32(false, a, false, bb,
                                                (short)0, acc, false, false);
  }
  #pragma unroll
  for (int r = 0; r < 8; ++r)
    out[((size_t)b * T + Mtile * 16 + r + hf * 8) * H + Hsub * 16 + m16] = acc[r];
}

// ---------------------------------------------------------------------------
// Host launcher. Inputs: x, pos_emb, W1, W2, b2, Wv (all fp32).
// Workspace layout (needs ~7.6 MB):
//   [0,1MB)   XK   fp32 [B*T][C]
//   [1,2MB)   XQ   fp32 [B*T][C]
//   [2,3MB)   V    fp32 [B*T][H]
//   [3,7MB)   WEI  fp32 [B][T][T]
//   [7MB..)   xb bf16 (512KB) | W1T bf16 (64KB) | WvT bf16 (32KB)
// ---------------------------------------------------------------------------
extern "C" void kernel_launch(void* const* d_in, const int* in_sizes, int n_in,
                              void* d_out, int out_size, void* d_ws, size_t ws_size,
                              hipStream_t stream) {
  const float* x       = (const float*)d_in[0];
  const float* pos_emb = (const float*)d_in[1];
  const float* W1      = (const float*)d_in[2];
  const float* W2      = (const float*)d_in[3];
  const float* b2      = (const float*)d_in[4];
  const float* Wv      = (const float*)d_in[5];
  float* out = (float*)d_out;

  char* ws = (char*)d_ws;
  float* XK  = (float*)(ws + 0u);
  float* XQ  = (float*)(ws + (1u << 20));
  float* Vv  = (float*)(ws + (2u << 20));
  float* WEI = (float*)(ws + (3u << 20));
  bf16_t* xb  = (bf16_t*)(ws + (7u << 20));
  bf16_t* W1T = (bf16_t*)(ws + (7u << 20) + 524288u);
  bf16_t* WvT = (bf16_t*)(ws + (7u << 20) + 524288u + 65536u);

  prep_convert<<<512, 256, 0, stream>>>(x, W1, Wv, xb, W1T, WvT);
  proj_gemm<<<dim3(128, 1, 3), 256, 0, stream>>>(xb, W1T, WvT, XK, XQ, Vv);
  fused_logits<<<dim3(T / TKT, T / TQT), 256, 0, stream>>>(pos_emb, W1T, XK, XQ,
                                                           W2, b2, WEI);
  softmax_rows<<<B * T, 256, 0, stream>>>(WEI);
  attn_out<<<128, 256, 0, stream>>>(WEI, Vv, out);
}
